// Mamba2MIL_66838281060833
// MI455X (gfx1250) — compile-verified, matrix operand order checked by reading, not run on hardware
//
#include <hip/hip_runtime.h>
#include <hip/hip_bf16.h>
#include <math.h>

// ---------------- Model dimensions (from the reference) ----------------
#define L_SEQ     4096
#define IN_DIM    1024
#define DMODEL    512
#define DSTATE    256
#define DCONV     4
#define DINNER    2048
#define HEADDIM   64
#define NHEADS    32
#define CONVDIM   2560            // DINNER + 2*DSTATE
#define DINPROJ   4640            // 2*DINNER + 2*DSTATE + NHEADS
#define NLAYERS   2
#define NCLS      4
#define EPSV      1e-5f

// ---------------- WMMA types (CDNA5, wave32) ----------------
typedef __attribute__((ext_vector_type(16))) __bf16 v16bf;
typedef __attribute__((ext_vector_type(8)))  float  v8f;

union Frag { v16bf v; uint4 q[2]; };

__device__ __forceinline__ unsigned short f32_to_bf16_rne(float f) {
  union { float f; unsigned int u; } x; x.f = f;
  unsigned int r = x.u + 0x7fffu + ((x.u >> 16) & 1u);
  return (unsigned short)(r >> 16);
}

// ---------------- f32 -> bf16 bulk conversion (4 elems / thread) -------
__global__ __launch_bounds__(256) void cvt_bf16_x4(
    const float* __restrict__ in, unsigned short* __restrict__ out, int n4)
{
  int i = blockIdx.x * 256 + threadIdx.x;
  if (i >= n4) return;
  float4 f = ((const float4*)in)[i];
  ushort4 o;
  o.x = f32_to_bf16_rne(f.x); o.y = f32_to_bf16_rne(f.y);
  o.z = f32_to_bf16_rne(f.z); o.w = f32_to_bf16_rne(f.w);
  ((ushort4*)out)[i] = o;
}

// =======================================================================
// GEMM:  out[M,N] = act( Xb[M,K](bf16) * Wb[N,K]^T(bf16) + bias ) (+res)
// Each wave owns a 32x32 macro-tile: 2 A-frags x 2 B-frags -> 4 WMMA per
// 8 x b128 loads. Block = 8 waves = 256M x 32N tile.
// Fragment layout (ISA 7.12.2, 16-bit A 16x32): lane<16 -> K {0..7,16..23},
// lane>=16 -> K {8..15,24..31}; B mirrors with N index (W rows are K-major).
// act: 0=none, 1=relu, 2=tanh
// =======================================================================
__device__ __forceinline__ void store_tile(
    v8f acc, int row0, int col, int half, const float* bias,
    const float* residual, float* __restrict__ out, int N, int act)
{
  const float bval = bias ? bias[col] : 0.f;
#pragma unroll
  for (int r = 0; r < 8; ++r) {
    const int row = row0 + half * 8 + r;
    float v = acc[r] + bval;
    if (act == 1)      v = fmaxf(v, 0.f);
    else if (act == 2) v = tanhf(v);
    if (residual) v += residual[(size_t)row * N + col];
    out[(size_t)row * N + col] = v;
  }
}

__global__ __launch_bounds__(256) void gemm_wmma_bf16(
    const unsigned short* __restrict__ Xb, const unsigned short* __restrict__ Wb,
    const float* __restrict__ bias, const float* __restrict__ residual,
    float* __restrict__ out, int M, int N, int K, int act)
{
  const int tid  = threadIdx.x;
  const int wave = tid >> 5;
  const int lane = tid & 31;
  const int half = lane >> 4;       // 0: lanes 0-15, 1: lanes 16-31
  const int l16  = lane & 15;
  const int mBase = blockIdx.y * 256 + wave * 32;
  const int nBase = blockIdx.x * 32;
  if (mBase >= M || nBase >= N) return;
  const int klo = half * 8;

  const unsigned short* a0p = Xb + (size_t)(mBase + l16) * K + klo;
  const unsigned short* a1p = a0p + (size_t)16 * K;
  const unsigned short* b0p = Wb + (size_t)(nBase + l16) * K + klo;
  const unsigned short* b1p = b0p + (size_t)16 * K;

  v8f c00 = {0.f,0.f,0.f,0.f,0.f,0.f,0.f,0.f};
  v8f c01 = c00, c10 = c00, c11 = c00;

  for (int kb = 0; kb < K; kb += 32) {
    if (kb + 256 < K) {
      __builtin_prefetch(a0p + kb + 256, 0, 1);
      __builtin_prefetch(b0p + kb + 256, 0, 1);
    }
    Frag a0, a1, b0, b1;
    a0.q[0] = *(const uint4*)(a0p + kb); a0.q[1] = *(const uint4*)(a0p + kb + 16);
    a1.q[0] = *(const uint4*)(a1p + kb); a1.q[1] = *(const uint4*)(a1p + kb + 16);
    b0.q[0] = *(const uint4*)(b0p + kb); b0.q[1] = *(const uint4*)(b0p + kb + 16);
    b1.q[0] = *(const uint4*)(b1p + kb); b1.q[1] = *(const uint4*)(b1p + kb + 16);
    c00 = __builtin_amdgcn_wmma_f32_16x16x32_bf16(false, a0.v, false, b0.v, (short)0, c00, false, false);
    c01 = __builtin_amdgcn_wmma_f32_16x16x32_bf16(false, a0.v, false, b1.v, (short)0, c01, false, false);
    c10 = __builtin_amdgcn_wmma_f32_16x16x32_bf16(false, a1.v, false, b0.v, (short)0, c10, false, false);
    c11 = __builtin_amdgcn_wmma_f32_16x16x32_bf16(false, a1.v, false, b1.v, (short)0, c11, false, false);
  }

  const int colBase = nBase + l16;
  store_tile(c00, mBase +  0, colBase +  0, half, bias, residual, out, N, act);
  store_tile(c01, mBase +  0, colBase + 16, half, bias, residual, out, N, act);
  store_tile(c10, mBase + 16, colBase +  0, half, bias, residual, out, N, act);
  store_tile(c11, mBase + 16, colBase + 16, half, bias, residual, out, N, act);
}

// ---------------- LayerNorm (one row / block) ----------------
__global__ __launch_bounds__(256) void layernorm_kernel(
    const float* __restrict__ in, const float* __restrict__ w,
    const float* __restrict__ b, float* __restrict__ out, int D)
{
  __shared__ float red[256];
  const int m = blockIdx.x, tid = threadIdx.x;
  const float* row = in + (size_t)m * D;
  float s = 0.f;
  for (int e = tid; e < D; e += 256) s += row[e];
  red[tid] = s; __syncthreads();
  for (int k = 128; k > 0; k >>= 1) { if (tid < k) red[tid] += red[tid + k]; __syncthreads(); }
  const float mean = red[0] / (float)D;
  __syncthreads();
  float v = 0.f;
  for (int e = tid; e < D; e += 256) { float d = row[e] - mean; v += d * d; }
  red[tid] = v; __syncthreads();
  for (int k = 128; k > 0; k >>= 1) { if (tid < k) red[tid] += red[tid + k]; __syncthreads(); }
  const float inv = rsqrtf(red[0] / (float)D + EPSV);
  for (int e = tid; e < D; e += 256)
    out[(size_t)m * D + e] = (row[e] - mean) * inv * w[e] + b[e];
}

// ---------------- dt = softplus(zx[:, -32:] + dt_bias) ----------------
__global__ void dt_softplus_kernel(const float* __restrict__ zx,
                                   const float* __restrict__ dtb,
                                   float* __restrict__ dt)
{
  int idx = blockIdx.x * blockDim.x + threadIdx.x;
  if (idx >= L_SEQ * NHEADS) return;
  int l = idx >> 5, h = idx & 31;
  float x = zx[(size_t)l * DINPROJ + (DINPROJ - NHEADS) + h] + dtb[h];
  dt[idx] = (x > 20.f) ? x : log1pf(expf(x));
}

// ---------------- causal depthwise conv (K=4) + SiLU ----------------
__global__ void conv_silu_kernel(const float* __restrict__ zx,
                                 const float* __restrict__ cw,
                                 const float* __restrict__ cb,
                                 float* __restrict__ xbc)
{
  int idx = blockIdx.x * blockDim.x + threadIdx.x;
  if (idx >= L_SEQ * CONVDIM) return;
  int l = idx / CONVDIM;
  int c = idx - l * CONVDIM;
  float a = cb[c];
#pragma unroll
  for (int k = 0; k < DCONV; ++k) {
    int ls = l + k - (DCONV - 1);
    if (ls >= 0) a += zx[(size_t)ls * DINPROJ + DINNER + c] * cw[c * DCONV + k];
  }
  xbc[idx] = a / (1.f + expf(-a));
}

// ---------------- sequential SSD scan: one block per head ----------------
// State h[64(p) x 256(n)] in registers: thread owns p = tid>>2, 64-wide n
// slice (tid&3)*64. B/C/x staged into LDS each step via the CDNA5 async
// global->LDS DMA path (ASYNCcnt), falling back to plain loads off-gfx1250.
__global__ __launch_bounds__(256) void ssd_scan_kernel(
    const float* __restrict__ dt, const float* __restrict__ xbc,
    const float* __restrict__ A_log, const float* __restrict__ Dvec,
    float* __restrict__ y)
{
  const int h = blockIdx.x;
  const int tid = threadIdx.x;
  const int p  = tid >> 2;
  const int q  = tid & 3;
  const int nb = q * 64;
  const float A  = -expf(A_log[h]);
  const float Dp = Dvec[h];

  __shared__ float sB[DSTATE], sC[DSTATE], sx[HEADDIM], sred[256];

  float st[64];
#pragma unroll
  for (int j = 0; j < 64; ++j) st[j] = 0.f;

  for (int t = 0; t < L_SEQ; ++t) {
    const float* row = xbc + (size_t)t * CONVDIM;
#if defined(__gfx1250__)
    {
      // LDS byte offset = low 32 bits of the generic shared-space address
      unsigned offB = (unsigned)(uintptr_t)(&sB[tid]);
      unsigned offC = (unsigned)(uintptr_t)(&sC[tid]);
      const float* gB = row + DINNER + tid;
      const float* gC = row + DINNER + DSTATE + tid;
      asm volatile("global_load_async_to_lds_b32 %0, %1, off"
                   :: "v"(offB), "v"(gB) : "memory");
      asm volatile("global_load_async_to_lds_b32 %0, %1, off"
                   :: "v"(offC), "v"(gC) : "memory");
      if (tid < HEADDIM) {
        unsigned offx = (unsigned)(uintptr_t)(&sx[tid]);
        const float* gx = row + h * HEADDIM + tid;
        asm volatile("global_load_async_to_lds_b32 %0, %1, off"
                     :: "v"(offx), "v"(gx) : "memory");
      }
      asm volatile("s_wait_asynccnt 0x0" ::: "memory");
    }
#else
    sB[tid] = row[DINNER + tid];
    sC[tid] = row[DINNER + DSTATE + tid];
    if (tid < HEADDIM) sx[tid] = row[h * HEADDIM + tid];
#endif
    __syncthreads();

    const float dtv   = dt[t * NHEADS + h];
    const float decay = expf(dtv * A);
    const float dtx   = dtv * sx[p];
    float acc = 0.f;
#pragma unroll
    for (int j = 0; j < 64; ++j) {
      st[j] = st[j] * decay + dtx * sB[nb + j];
      acc  += st[j] * sC[nb + j];
    }
    sred[tid] = acc;
    __syncthreads();
    if (q == 0) {
      float r = sred[tid] + sred[tid + 1] + sred[tid + 2] + sred[tid + 3];
      y[(size_t)t * DINNER + h * HEADDIM + p] = r + Dp * sx[p];
    }
    __syncthreads();
  }
}

// ---------------- y = rmsnorm(y * silu(z)) * rms_w ----------------
__global__ __launch_bounds__(256) void gate_rms_kernel(
    float* __restrict__ y, const float* __restrict__ zx,
    const float* __restrict__ rw)
{
  __shared__ float red[256];
  const int m = blockIdx.x, tid = threadIdx.x;
  float ss = 0.f;
  for (int e = tid; e < DINNER; e += 256) {
    float z = zx[(size_t)m * DINPROJ + e];
    float g = y[(size_t)m * DINNER + e] * (z / (1.f + expf(-z)));
    y[(size_t)m * DINNER + e] = g;
    ss += g * g;
  }
  red[tid] = ss; __syncthreads();
  for (int k = 128; k > 0; k >>= 1) { if (tid < k) red[tid] += red[tid + k]; __syncthreads(); }
  const float scale = rsqrtf(red[0] / (float)DINNER + EPSV);
  for (int e = tid; e < DINNER; e += 256)
    y[(size_t)m * DINNER + e] *= scale * rw[e];
}

// ---------------- a_raw[l] = tanh-feats @ att_w2 + b2 ----------------
__global__ void att_score_kernel(const float* __restrict__ t,
                                 const float* __restrict__ w2,
                                 const float* __restrict__ b2,
                                 float* __restrict__ a)
{
  int l = blockIdx.x * blockDim.x + threadIdx.x;
  if (l >= L_SEQ) return;
  float acc = b2[0];
  for (int j = 0; j < 128; ++j) acc += t[(size_t)l * 128 + j] * w2[j];
  a[l] = acc;
}

// ---------------- softmax over L (single block) ----------------
__global__ __launch_bounds__(256) void softmax_kernel(float* __restrict__ a)
{
  __shared__ float red[256];
  const int tid = threadIdx.x;
  float mx = -3.4e38f;
  for (int l = tid; l < L_SEQ; l += 256) mx = fmaxf(mx, a[l]);
  red[tid] = mx; __syncthreads();
  for (int k = 128; k > 0; k >>= 1) { if (tid < k) red[tid] = fmaxf(red[tid], red[tid + k]); __syncthreads(); }
  const float M = red[0]; __syncthreads();
  float s = 0.f;
  for (int l = tid; l < L_SEQ; l += 256) s += expf(a[l] - M);
  red[tid] = s; __syncthreads();
  for (int k = 128; k > 0; k >>= 1) { if (tid < k) red[tid] += red[tid + k]; __syncthreads(); }
  const float inv = 1.f / red[0];
  for (int l = tid; l < L_SEQ; l += 256) a[l] = expf(a[l] - M) * inv;
}

// ---------------- pooled[d] = sum_l a[l] * h[l,d] ----------------
__global__ __launch_bounds__(256) void pool_kernel(
    const float* __restrict__ a, const float* __restrict__ h,
    float* __restrict__ pooled)
{
  __shared__ float red[256];
  const int d = blockIdx.x, tid = threadIdx.x;
  float s = 0.f;
  for (int l = tid; l < L_SEQ; l += 256) s += a[l] * h[(size_t)l * DMODEL + d];
  red[tid] = s; __syncthreads();
  for (int k = 128; k > 0; k >>= 1) { if (tid < k) red[tid] += red[tid + k]; __syncthreads(); }
  if (tid == 0) pooled[d] = red[0];
}

// ---------------- classifier + softmax + argmax -> d_out[9] ----------------
__global__ void cls_kernel(const float* __restrict__ pooled,
                           const float* __restrict__ cw,
                           const float* __restrict__ cb,
                           float* __restrict__ out)
{
  __shared__ float lg[NCLS];
  const int tid = threadIdx.x;
  if (tid < NCLS) {
    float acc = cb[tid];
    for (int d = 0; d < DMODEL; ++d) acc += pooled[d] * cw[tid * DMODEL + d];
    lg[tid] = acc;
  }
  __syncthreads();
  if (tid == 0) {
    float m = lg[0]; int am = 0;
    for (int c = 1; c < NCLS; ++c) if (lg[c] > m) { m = lg[c]; am = c; }
    float e[NCLS], s = 0.f;
    for (int c = 0; c < NCLS; ++c) { e[c] = expf(lg[c] - m); s += e[c]; }
    for (int c = 0; c < NCLS; ++c) { out[c] = lg[c]; out[NCLS + c] = e[c] / s; }
    out[2 * NCLS] = (float)am;
  }
}

// =======================================================================
static inline void launch_cvt(const float* in, unsigned short* out, size_t n,
                              hipStream_t stream)
{
  int n4 = (int)(n / 4);
  cvt_bf16_x4<<<(n4 + 255) / 256, 256, 0, stream>>>(in, out, n4);
}

extern "C" void kernel_launch(void* const* d_in, const int* in_sizes, int n_in,
                              void* d_out, int out_size, void* d_ws, size_t ws_size,
                              hipStream_t stream)
{
  const float* x       = (const float*)d_in[0];
  const float* fc1_w   = (const float*)d_in[1];
  const float* fc1_b   = (const float*)d_in[2];
  const float* ln_w    = (const float*)d_in[3];
  const float* ln_b    = (const float*)d_in[4];
  const float* in_proj = (const float*)d_in[5];
  const float* conv_w  = (const float*)d_in[6];
  const float* conv_b  = (const float*)d_in[7];
  const float* dt_bias = (const float*)d_in[8];
  const float* A_log   = (const float*)d_in[9];
  const float* Dvec    = (const float*)d_in[10];
  const float* rms_w   = (const float*)d_in[11];
  const float* out_w   = (const float*)d_in[12];
  const float* norm_w  = (const float*)d_in[13];
  const float* norm_b  = (const float*)d_in[14];
  const float* att_w1  = (const float*)d_in[15];
  const float* att_b1  = (const float*)d_in[16];
  const float* att_w2  = (const float*)d_in[17];
  const float* att_b2  = (const float*)d_in[18];
  const float* cls_w   = (const float*)d_in[19];
  const float* cls_b   = (const float*)d_in[20];
  float* out = (float*)d_out;

  // ---------------- workspace layout (f32 units), ~185 MB ----------------
  float* ws = (float*)d_ws;
  size_t off = 0;
  float* h      = ws + off; off += (size_t)L_SEQ * DMODEL;
  float* hn     = ws + off; off += (size_t)L_SEQ * DMODEL;
  float* zx     = ws + off; off += (size_t)L_SEQ * DINPROJ;
  float* xbc    = ws + off; off += (size_t)L_SEQ * CONVDIM;
  float* dts    = ws + off; off += (size_t)L_SEQ * NHEADS;
  float* ybuf   = ws + off; off += (size_t)L_SEQ * DINNER;
  float* att1   = ws + off; off += (size_t)L_SEQ * 128;
  float* arow   = ws + off; off += (size_t)L_SEQ;
  float* pooled = ws + off; off += (size_t)DMODEL;
  off = (off + 63) & ~(size_t)63;
  unsigned short* actb = (unsigned short*)(ws + off); off += (size_t)L_SEQ * DINNER / 2;
  unsigned short* wb   = (unsigned short*)(ws + off); off += (size_t)DINPROJ * DMODEL / 2 + 64;

  const dim3 blk(256);

  // h = relu(x @ fc1_w^T + fc1_b)
  launch_cvt(x,     actb, (size_t)L_SEQ * IN_DIM,  stream);
  launch_cvt(fc1_w, wb,   (size_t)DMODEL * IN_DIM, stream);
  gemm_wmma_bf16<<<dim3(DMODEL / 32, L_SEQ / 256), blk, 0, stream>>>(
      actb, wb, fc1_b, nullptr, h, L_SEQ, DMODEL, IN_DIM, 1);

  for (int i = 0; i < NLAYERS; ++i) {
    layernorm_kernel<<<L_SEQ, blk, 0, stream>>>(
        h, ln_w + i * DMODEL, ln_b + i * DMODEL, hn, DMODEL);

    launch_cvt(hn, actb, (size_t)L_SEQ * DMODEL, stream);
    launch_cvt(in_proj + (size_t)i * DINPROJ * DMODEL, wb,
               (size_t)DINPROJ * DMODEL, stream);
    gemm_wmma_bf16<<<dim3(DINPROJ / 32, L_SEQ / 256), blk, 0, stream>>>(
        actb, wb, nullptr, nullptr, zx, L_SEQ, DINPROJ, DMODEL, 0);

    dt_softplus_kernel<<<(L_SEQ * NHEADS + 255) / 256, blk, 0, stream>>>(
        zx, dt_bias + i * NHEADS, dts);

    conv_silu_kernel<<<(L_SEQ * CONVDIM + 255) / 256, blk, 0, stream>>>(
        zx, conv_w + (size_t)i * CONVDIM * DCONV, conv_b + i * CONVDIM, xbc);

    ssd_scan_kernel<<<NHEADS, blk, 0, stream>>>(
        dts, xbc, A_log + i * NHEADS, Dvec + i * NHEADS, ybuf);

    gate_rms_kernel<<<L_SEQ, blk, 0, stream>>>(ybuf, zx, rms_w + i * DINNER);

    // h += y @ out_w^T
    launch_cvt(ybuf, actb, (size_t)L_SEQ * DINNER, stream);
    launch_cvt(out_w + (size_t)i * DMODEL * DINNER, wb,
               (size_t)DMODEL * DINNER, stream);
    gemm_wmma_bf16<<<dim3(DMODEL / 32, L_SEQ / 256), blk, 0, stream>>>(
        actb, wb, nullptr, h, h, L_SEQ, DMODEL, DINNER, 0);
  }

  layernorm_kernel<<<L_SEQ, blk, 0, stream>>>(h, norm_w, norm_b, hn, DMODEL);

  launch_cvt(hn,     actb, (size_t)L_SEQ * DMODEL, stream);
  launch_cvt(att_w1, wb,   (size_t)128 * DMODEL,   stream);
  gemm_wmma_bf16<<<dim3(128 / 32, L_SEQ / 256), blk, 0, stream>>>(
      actb, wb, att_b1, nullptr, att1, L_SEQ, 128, DMODEL, 2);

  att_score_kernel<<<L_SEQ / 256, blk, 0, stream>>>(att1, att_w2, att_b2, arow);
  softmax_kernel<<<1, blk, 0, stream>>>(arow);
  pool_kernel<<<DMODEL, blk, 0, stream>>>(arow, hn, pooled);
  cls_kernel<<<1, 64, 0, stream>>>(pooled, cls_w, cls_b, out);
}
